// WindowAttention_55336358642632
// MI455X (gfx1250) — matrix operand order repaired and checked by reading, not compiled, not run on hardware
//
#include <hip/hip_runtime.h>
#include <hip/hip_bf16.h>

typedef __attribute__((ext_vector_type(16))) _Float16 v16h;
typedef __attribute__((ext_vector_type(8)))  float    v8f;
typedef __attribute__((ext_vector_type(4)))  unsigned int v4u;
typedef __attribute__((ext_vector_type(8)))  int      v8i;
typedef __attribute__((ext_vector_type(4)))  int      v4i;

#define CD   256        // channels
#define XS   272        // row stride (halves) for X/Q/K/O LDS buffers (256 + 16 pad)
#define VS   72         // row stride (halves) for transposed-V buffer (64 + 8 pad)
#define AS   72         // row stride (halves) for per-wave softmax(P) buffer
// LDS layout (halves): Xs[64*XS] Qs[64*XS] Ks[64*XS] Vt[256*VS] A[8 waves][64*AS]
#define SMEM_BYTES ((64*XS*3 + 256*VS + 8*64*AS) * 2)   // = 215040 bytes

static __device__ __forceinline__ v8f wmma_f16(v16h a, v16h b, v8f c) {
  // D = A(16x32 f16) * B(32x16 f16) + C(16x16 f32)
  return __builtin_amdgcn_wmma_f32_16x16x32_f16(
      /*neg_a=*/false, a, /*neg_b=*/false, b,
      /*c_mod=*/(short)0, c, /*reuse_a=*/false, /*reuse_b=*/false);
}

// A fragment: rows m0..m0+15, k0..k0+31, from row-major f16 (stride in halves).
// gfx1250 layout: lanes 0-15 hold K {0-7,16-23}, lanes 16-31 hold K {8-15,24-31}.
static __device__ __forceinline__ v16h load_frag_A(const _Float16* base, int m0,
                                                   int k0, int stride, int lane) {
  int m  = m0 + (lane & 15);
  int kb = (lane < 16) ? 0 : 8;
  const _Float16* p = base + m * stride + k0 + kb;
  v16h r;
  ((uint4*)&r)[0] = *(const uint4*)(p);        // halves 0-7  -> K = k0+kb .. +7
  ((uint4*)&r)[1] = *(const uint4*)(p + 16);   // halves 8-15 -> K = k0+16+kb ..
  return r;
}

// B fragment: B[k][n] = src[n][k] (src row-major, k contiguous within a row).
// gfx1250 layout: lanes 0-15 hold K {0-7,16-23}, lanes 16-31 hold K {8-15,24-31}, N = lane%16.
static __device__ __forceinline__ v16h load_frag_B(const _Float16* base, int n0,
                                                   int k0, int stride, int lane) {
  int n  = n0 + (lane & 15);
  int kb = (lane < 16) ? 0 : 8;
  const _Float16* p = base + n * stride + k0 + kb;
  v16h r;
  ((uint4*)&r)[0] = *(const uint4*)(p);
  ((uint4*)&r)[1] = *(const uint4*)(p + 16);
  return r;
}

// ---------------------------------------------------------------------------
// Weight pre-conversion: Wq,Wk,Wv,Wo (each 256x256 f32) -> f16 in scratch.
// ---------------------------------------------------------------------------
__global__ void convert_weights(const float* __restrict__ wq,
                                const float* __restrict__ wk,
                                const float* __restrict__ wv,
                                const float* __restrict__ wo,
                                _Float16* __restrict__ w16) {
  int i = blockIdx.x * 256 + threadIdx.x;          // 65536 threads total
  w16[i            ] = (_Float16)wq[i];
  w16[i +   65536  ] = (_Float16)wk[i];
  w16[i + 2*65536  ] = (_Float16)wv[i];
  w16[i + 3*65536  ] = (_Float16)wo[i];
}

// ---------------------------------------------------------------------------
// One workgroup = one 8x8 window. 8 waves; wave w owns head w (channels 32w..).
// ---------------------------------------------------------------------------
__global__ __launch_bounds__(256, 1)
void swin_window_attn(const float* __restrict__ x,        // [4][256][256][256]
                      const _Float16* __restrict__ w16,   // [4][256][256] q,k,v,o
                      const float* __restrict__ bias_table, // [8][225]
                      const int* __restrict__ rel,        // [64][64]
                      const float* __restrict__ bo,       // [256]
                      float* __restrict__ out) {          // [4][256][256][256]
  extern __shared__ char smem[];
  _Float16* Xs = (_Float16*)smem;            // 64 x XS  (X, later reused for O)
  _Float16* Qs = Xs + 64 * XS;               // 64 x XS
  _Float16* Ks = Qs + 64 * XS;               // 64 x XS
  _Float16* Vt = Ks + 64 * XS;               // 256 x VS  (V transposed: [chan][tok])
  _Float16* Ab = Vt + 256 * VS;              // 8 x (64 x AS)

  const int tid  = threadIdx.x;
  const int wave = tid >> 5;
  const int lane = tid & 31;
  const int nloc = lane & 15;
  const int hl   = (lane < 16) ? 0 : 8;
  _Float16* Aw = Ab + wave * (64 * AS);

  const int blk = blockIdx.x;                // 4096 = 4 * 32 * 32
  const int b  = blk >> 10;
  const int wh = (blk >> 5) & 31;
  const int ww = blk & 31;

  const float* xwin = x + ((size_t)b * CD) * 65536 + (size_t)(wh * 8) * 256 + ww * 8;

  // ---- Stage 1: window (64x256 f32) -> LDS, then f32->f16 token-major -----
#if __has_builtin(__builtin_amdgcn_tensor_load_to_lds)
  {
    // TDM: 3D tile (x=8 cols, y=8 rows, z=256 channels) -> contiguous LDS
    // staging [c][i][j] (f32, 64KB), aliasing Qs/Ks which are written later.
    float* Xstage = (float*)Qs;
    unsigned long long gaddr = (unsigned long long)(uintptr_t)xwin;
    unsigned ldsoff = (unsigned)(uintptr_t)(void*)Xstage;  // low 32b = LDS offset
    if (wave == 0) {
      // D# group0: count=1, lds_addr, global_addr[56:0], type=2 ("image")
      v4u g0 = { 1u, ldsoff, (unsigned)gaddr,
                 (unsigned)(gaddr >> 32) | (2u << 30) };
      // D# group1: data_size=4B(2<<16); tensor_dim0=256 (bits79:48);
      // tensor_dim1=256 (bits111:80); tile_dim0=8 (bits127:112);
      // tile_dim1=8, tile_dim2=256 (bits143:128,159:144);
      // dim0_stride=256 (bits207:160); dim1_stride=65536 (bits255:208)
      v8i g1 = { 0x20000, 0x01000000, 0x01000000, 0x00080000,
                 0x01000008, 256, 0, 1 };
      // D# group2: tensor_dim2=256; tile_dim3=0 (unused)
      v4i g2 = { 256, 0, 0, 0 };
      v4i g3 = { 0, 0, 0, 0 };
#if defined(__clang_major__) && (__clang_major__ >= 23)
      __builtin_amdgcn_tensor_load_to_lds(g0, g1, g2, g3, (v8i)0, 0);
#else
      __builtin_amdgcn_tensor_load_to_lds(g0, g1, g2, g3, 0);
#endif
#if __has_builtin(__builtin_amdgcn_s_wait_tensorcnt)
      __builtin_amdgcn_s_wait_tensorcnt((short)0);
#else
      asm volatile("s_wait_tensorcnt 0x0" ::: "memory");
#endif
    }
    __syncthreads();
    // transpose-convert: Xstage[c*64 + l] (f32) -> Xs[l*XS + c] (f16)
    for (int it = 0; it < 64; ++it) {
      int flat = it * 256 + tid;             // contiguous LDS reads
      int c = flat >> 6, l = flat & 63;
      Xs[l * XS + c] = (_Float16)Xstage[flat];
    }
  }
#else
  for (int it = 0; it < 64; ++it) {
    int flat = it * 256 + tid;               // 64*256 = 16384 elements
    int c = flat >> 6, l = flat & 63;
    float v = xwin[(size_t)c * 65536 + (l >> 3) * 256 + (l & 7)];
    Xs[l * XS + c] = (_Float16)v;
  }
#endif
  __syncthreads();

  // ---- Stage 2: Q,K,V = X * W^T; wave w computes output cols [32w,32w+32) --
  for (int mat = 0; mat < 3; ++mat) {
    const _Float16* Wm = w16 + (size_t)mat * 65536;
    for (int tn = 0; tn < 2; ++tn) {
      int n0 = wave * 32 + tn * 16;
      __builtin_prefetch(Wm + (size_t)n0 * 256, 0, 1);  // global_prefetch_b8
      v8f acc[4] = {};
      for (int ks = 0; ks < 8; ++ks) {
        v16h bf = load_frag_B(Wm, n0, ks * 32, 256, lane);
        #pragma unroll
        for (int mt = 0; mt < 4; ++mt) {
          v16h af = load_frag_A(Xs, mt * 16, ks * 32, XS, lane);
          acc[mt] = wmma_f16(af, bf, acc[mt]);
        }
      }
      int n = n0 + nloc;
      #pragma unroll
      for (int mt = 0; mt < 4; ++mt) {
        int mb = mt * 16 + hl;
        #pragma unroll
        for (int r = 0; r < 8; ++r) {
          _Float16 hv = (_Float16)acc[mt][r];
          int m = mb + r;
          if (mat == 0)      Qs[m * XS + n] = hv;
          else if (mat == 1) Ks[m * XS + n] = hv;
          else               Vt[n * VS + m] = hv;   // transposed store
        }
      }
    }
  }
  __syncthreads();

  // ---- Stage 3: per-head attention (head = wave) --------------------------
  const int h = wave;
  v8f s[4][4];
  #pragma unroll
  for (int mt = 0; mt < 4; ++mt) {
    v16h qf = load_frag_A(Qs, mt * 16, h * 32, XS, lane);  // K = 32 = head dim
    #pragma unroll
    for (int tn = 0; tn < 4; ++tn) {
      v16h kf = load_frag_B(Ks, tn * 16, h * 32, XS, lane);
      v8f z = {};
      s[mt][tn] = wmma_f16(qf, kf, z);
    }
  }
  const float rscale = 0.1767766952966369f;   // 1/sqrt(32)
  #pragma unroll
  for (int mt = 0; mt < 4; ++mt) {
    #pragma unroll
    for (int r = 0; r < 8; ++r) {
      int m = mt * 16 + hl + r;               // query token (this lane's row)
      float tmp[4];
      float vmax = -3.0e38f;
      #pragma unroll
      for (int tn = 0; tn < 4; ++tn) {
        int n = tn * 16 + nloc;               // key token
        float val = s[mt][tn][r] * rscale + bias_table[h * 225 + rel[m * 64 + n]];
        tmp[tn] = val;
        vmax = fmaxf(vmax, val);
      }
      // row reduction: columns live in 16-lane halves (+ 4 in-lane tiles)
      vmax = fmaxf(vmax, __shfl_xor(vmax, 8, 32));
      vmax = fmaxf(vmax, __shfl_xor(vmax, 4, 32));
      vmax = fmaxf(vmax, __shfl_xor(vmax, 2, 32));
      vmax = fmaxf(vmax, __shfl_xor(vmax, 1, 32));
      float sum = 0.f;
      #pragma unroll
      for (int tn = 0; tn < 4; ++tn) { tmp[tn] = __expf(tmp[tn] - vmax); sum += tmp[tn]; }
      sum += __shfl_xor(sum, 8, 32);
      sum += __shfl_xor(sum, 4, 32);
      sum += __shfl_xor(sum, 2, 32);
      sum += __shfl_xor(sum, 1, 32);
      float inv = 1.0f / sum;
      #pragma unroll
      for (int tn = 0; tn < 4; ++tn)
        Aw[m * AS + tn * 16 + nloc] = (_Float16)(tmp[tn] * inv);
    }
  }
  // O_head = P * V  -> into Xs columns [32h, 32h+32)
  #pragma unroll
  for (int mt = 0; mt < 4; ++mt) {
    #pragma unroll
    for (int tn = 0; tn < 2; ++tn) {
      v8f acc = {};
      #pragma unroll
      for (int ks = 0; ks < 2; ++ks) {        // K = 64 keys = 2 WMMA steps
        v16h af = load_frag_A(Aw, mt * 16, ks * 32, AS, lane);
        v16h vf = load_frag_B(Vt, h * 32 + tn * 16, ks * 32, VS, lane);
        acc = wmma_f16(af, vf, acc);
      }
      int n  = h * 32 + tn * 16 + nloc;
      int mb = mt * 16 + hl;
      #pragma unroll
      for (int r = 0; r < 8; ++r)
        Xs[(mb + r) * XS + n] = (_Float16)acc[r];
    }
  }
  __syncthreads();

  // ---- Stage 4: output projection + bias, scatter to NCHW -----------------
  const _Float16* Wo16 = w16 + (size_t)3 * 65536;
  const size_t obase = ((size_t)b * CD) * 65536 + (size_t)(wh * 8) * 256 + ww * 8;
  for (int tn = 0; tn < 2; ++tn) {
    int n0 = wave * 32 + tn * 16;
    __builtin_prefetch(Wo16 + (size_t)n0 * 256, 0, 1);
    v8f acc[4] = {};
    for (int ks = 0; ks < 8; ++ks) {
      v16h bf = load_frag_B(Wo16, n0, ks * 32, 256, lane);
      #pragma unroll
      for (int mt = 0; mt < 4; ++mt) {
        v16h af = load_frag_A(Xs, mt * 16, ks * 32, XS, lane);
        acc[mt] = wmma_f16(af, bf, acc[mt]);
      }
    }
    int n = n0 + nloc;
    float bb = bo[n];
    #pragma unroll
    for (int mt = 0; mt < 4; ++mt) {
      int mb = mt * 16 + hl;
      #pragma unroll
      for (int r = 0; r < 8; ++r) {
        int m = mb + r;
        out[obase + (size_t)n * 65536 + (m >> 3) * 256 + (m & 7)] = acc[mt][r] + bb;
      }
    }
  }
}

extern "C" void kernel_launch(void* const* d_in, const int* in_sizes, int n_in,
                              void* d_out, int out_size, void* d_ws, size_t ws_size,
                              hipStream_t stream) {
  const float* x          = (const float*)d_in[0];
  const float* Wq         = (const float*)d_in[1];
  const float* Wk         = (const float*)d_in[2];
  const float* Wv         = (const float*)d_in[3];
  const float* bias_table = (const float*)d_in[4];
  const float* Wo         = (const float*)d_in[5];
  const float* bo         = (const float*)d_in[6];
  const int*   rel        = (const int*)d_in[7];
  float* out = (float*)d_out;

  _Float16* w16 = (_Float16*)d_ws;   // needs 4*65536*2 = 512 KB of scratch

  convert_weights<<<256, 256, 0, stream>>>(Wq, Wk, Wv, Wo, w16);

  (void)hipFuncSetAttribute(reinterpret_cast<const void*>(swin_window_attn),
                            hipFuncAttributeMaxDynamicSharedMemorySize, SMEM_BYTES);
  swin_window_attn<<<4096, 256, SMEM_BYTES, stream>>>(x, w16, bias_table, rel, bo, out);
}
